// StaticAddAttention_54485955117827
// MI455X (gfx1250) — matrix operand order repaired and checked by reading, not compiled, not run on hardware
//
#include <hip/hip_runtime.h>
#include <hip/hip_bf16.h>

// ---------------------------------------------------------------------------
// StaticAddAttention for MI455X (gfx1250, wave32, WMMA bf16)
//   T=256, B=8, Din=Dm=256, A=128
// Pipeline:
//   prep   : f32->bf16 staging (X, M, Wi^T, Wm^T, V^T) + concat half of out
//   proj   : ip = X@Wi, mp = M@Wm            (v_wmma_f32_16x16x32_bf16)
//   logits : logits[s,t,b] = sum_a tanh(ip+mp)*v   (native v_tanh, a-split x4)
//   softctx: masked softmax + context = P@V  (async-to-LDS stage + WMMA bf16)
// ---------------------------------------------------------------------------

#define T_DIM   256
#define B_DIM   8
#define D_DIM   256   // Din == Dm
#define A_DIM   128
#define NEG_INF (-1e30f)

typedef __attribute__((ext_vector_type(16))) __bf16 bf16x16;
typedef __attribute__((ext_vector_type(8)))  float  f32x8;
typedef __attribute__((ext_vector_type(4)))  int    i32x4;

union ABv {
    bf16x16 v;
    uint4   q[2];
};

__device__ __forceinline__ float fast_tanh(float x) {
#if __has_builtin(__builtin_amdgcn_tanhf)
    return __builtin_amdgcn_tanhf(x);      // native v_tanh_f32 on gfx1250
#else
    return tanhf(x);
#endif
}

__device__ __forceinline__ float fast_exp(float x) {
#if __has_builtin(__builtin_amdgcn_exp2f)
    return __builtin_amdgcn_exp2f(x * 1.4426950408889634f);
#else
    return __expf(x);
#endif
}

// Async global->LDS b128 copy (ASYNCcnt path) with plain-load fallback.
// Builtin params are int4 pointers (global / LDS address spaces); pass generic
// int4* and let clang perform the address-space conversions.
__device__ __forceinline__ void async_copy_b128(const float* gsrc, float* ldst) {
#if __has_builtin(__builtin_amdgcn_global_load_async_to_lds_b128)
    __builtin_amdgcn_global_load_async_to_lds_b128((i32x4*)gsrc, (i32x4*)ldst,
                                                   0, 0);
#else
    *(float4*)ldst = *(const float4*)gsrc;
#endif
}

__device__ __forceinline__ void async_copy_wait() {
#if __has_builtin(__builtin_amdgcn_global_load_async_to_lds_b128)
#if __has_builtin(__builtin_amdgcn_s_wait_asynccnt)
    __builtin_amdgcn_s_wait_asynccnt(0);
#else
    asm volatile("s_wait_asynccnt 0" ::: "memory");
#endif
#endif
}

// ---------------------------------------------------------------------------
// Kernel 0: precision conversion / layout prep + concat-copy half of output.
// ---------------------------------------------------------------------------
__global__ void prep_kernel(const float* __restrict__ input,
                            const float* __restrict__ memory,
                            const float* __restrict__ Wi,
                            const float* __restrict__ Wm,
                            float* __restrict__ out,
                            __bf16* __restrict__ Xbf,
                            __bf16* __restrict__ Mbf,
                            __bf16* __restrict__ WiT,
                            __bf16* __restrict__ WmT,
                            __bf16* __restrict__ VT) {
    const int stride = gridDim.x * blockDim.x;
    const int tid0   = blockIdx.x * blockDim.x + threadIdx.x;

    for (int i = tid0; i < T_DIM * B_DIM * D_DIM; i += stride) {
        float xi = input[i];
        Xbf[i] = (__bf16)xi;
        Mbf[i] = (__bf16)memory[i];
        int n = i >> 8;                        // n = t*B + b   (D=256)
        int d = i & (D_DIM - 1);
        out[n * (2 * D_DIM) + D_DIM + d] = xi; // concat half
        int t = n >> 3;                        // B = 8
        int b = n & (B_DIM - 1);
        VT[((b << 8) + d) * T_DIM + t] = (__bf16)xi;
    }

    for (int i = tid0; i < D_DIM * A_DIM; i += stride) {
        int d = i >> 7;                        // A = 128
        int a = i & (A_DIM - 1);
        WiT[a * D_DIM + d] = (__bf16)Wi[i];
        WmT[a * D_DIM + d] = (__bf16)Wm[i];
    }
}

// ---------------------------------------------------------------------------
// Kernel 1: projections via v_wmma_f32_16x16x32_bf16. One 16x16 tile/wave.
// ---------------------------------------------------------------------------
__global__ __launch_bounds__(128)
void proj_kernel(const __bf16* __restrict__ Xbf,
                 const __bf16* __restrict__ Mbf,
                 const __bf16* __restrict__ WiT,
                 const __bf16* __restrict__ WmT,
                 float* __restrict__ ipf,
                 float* __restrict__ mpf) {
    const int lane = threadIdx.x & 31;
    const int wave = threadIdx.x >> 5;
    const int tile = blockIdx.x * 4 + wave;      // 0..2047
    const int mat  = tile >> 10;                 // 0 -> ip, 1 -> mp
    const int rem  = tile & 1023;
    const int mt   = rem >> 3;                   // 0..127
    const int nt   = rem & 7;                    // 0..7

    const __bf16* __restrict__ Asrc = mat ? Mbf : Xbf;
    const __bf16* __restrict__ Bsrc = mat ? WmT : WiT;
    float*        __restrict__ Osrc = mat ? mpf : ipf;

    // ISA 16-bit A layout: lanes 0-15 hold K {0..7,16..23}; lanes 16-31 {8..15,24..31}
    const int arow   = mt * 16 + (lane & 15);
    const int abase0 = (lane < 16) ? 0 : 8;
    const int abase1 = abase0 + 16;
    // ISA 16-bit B layout: lane = col N, K {0..7,16..23} / {8..15,24..31}
    const int brow  = nt * 16 + (lane & 15);
    const int bbase = (lane < 16) ? 0 : 8;

    f32x8 acc = {};
    #pragma unroll
    for (int k0 = 0; k0 < D_DIM; k0 += 32) {
        ABv a, b;
        a.q[0] = *(const uint4*)(Asrc + arow * D_DIM + k0 + abase0);
        a.q[1] = *(const uint4*)(Asrc + arow * D_DIM + k0 + abase1);
        b.q[0] = *(const uint4*)(Bsrc + brow * D_DIM + k0 + bbase);
        b.q[1] = *(const uint4*)(Bsrc + brow * D_DIM + k0 + bbase + 16);
        acc = __builtin_amdgcn_wmma_f32_16x16x32_bf16(
            false, a.v, false, b.v, (short)0, acc, false, false);
    }

    const int mrow0 = mt * 16 + ((lane < 16) ? 0 : 8);
    const int ncol  = nt * 16 + (lane & 15);
    #pragma unroll
    for (int r = 0; r < 8; ++r)
        Osrc[(mrow0 + r) * A_DIM + ncol] = acc[r];
}

// ---------------------------------------------------------------------------
// Kernel 2a: masked logits. Block = (s-tile 16, t-tile 64, b); 256 threads =
// 64 t x 4 a-quarters (a-split quadruples wave-level parallelism of the
// trans-op-bound loop). Partials reduced through LDS, written to Lg.
//   Lg[((stile*8+b)*16 + s)*256 + t]
// ---------------------------------------------------------------------------
__global__ __launch_bounds__(256)
void logits_kernel(const float* __restrict__ ipf,
                   const float* __restrict__ mpf,
                   const int* __restrict__ mask,
                   const float* __restrict__ v,
                   float* __restrict__ Lg) {
    __shared__ __align__(16) float s_ip[A_DIM * 16];     // [a][s] transposed
    __shared__ __align__(16) float s_v[A_DIM];
    __shared__ __align__(16) float s_part[256 * 16];     // [q*64+tq][s]

    const int tid   = threadIdx.x;
    const int blk   = blockIdx.x;
    const int b     = blk & 7;
    const int ttile = (blk >> 3) & 3;    // 4 t-tiles of 64
    const int stile = blk >> 5;          // 16 s-tiles of 16
    const int sbase = stile * 16;

    const int tq    = tid & 63;
    const int q     = tid >> 6;          // a-quarter 0..3
    const int t     = ttile * 64 + tq;
    const int abase = q * 32;

    for (int i = tid; i < A_DIM * 16; i += 256) {
        int a = i >> 4, s = i & 15;
        s_ip[i] = ipf[((sbase + s) * B_DIM + b) * A_DIM + a];
    }
    if (tid < A_DIM) s_v[tid] = v[tid];
    __syncthreads();

    const float* __restrict__ mprow = mpf + (t * B_DIM + b) * A_DIM;
    __builtin_prefetch(mprow + abase, 0, 1);    // global_prefetch_b8
    float acc[16];
    #pragma unroll
    for (int s = 0; s < 16; ++s) acc[s] = 0.0f;

    for (int a = abase; a < abase + 32; a += 4) {
        float4 m4 = *(const float4*)(mprow + a);
        float4 v4 = *(const float4*)(s_v + a);
        #pragma unroll
        for (int j = 0; j < 4; ++j) {
            float mv = (&m4.x)[j];
            float vv = (&v4.x)[j];
            const float4* iprow = (const float4*)(s_ip + (a + j) * 16);
            #pragma unroll
            for (int qq = 0; qq < 4; ++qq) {
                float4 ip4 = iprow[qq];          // broadcast LDS read
                acc[qq * 4 + 0] += fast_tanh(ip4.x + mv) * vv;
                acc[qq * 4 + 1] += fast_tanh(ip4.y + mv) * vv;
                acc[qq * 4 + 2] += fast_tanh(ip4.z + mv) * vv;
                acc[qq * 4 + 3] += fast_tanh(ip4.w + mv) * vv;
            }
        }
    }
    #pragma unroll
    for (int s = 0; s < 16; ++s) s_part[tid * 16 + s] = acc[s];
    __syncthreads();

    // combine the 4 a-quarters, apply mask, write 64x16 logits
    for (int idx = tid; idx < 64 * 16; idx += 256) {
        const int s   = idx >> 6;
        const int tq2 = idx & 63;
        float sum = 0.0f;
        #pragma unroll
        for (int qq = 0; qq < 4; ++qq)
            sum += s_part[(qq * 64 + tq2) * 16 + s];
        const int t2 = ttile * 64 + tq2;
        const bool ok = (mask[t2 * B_DIM + b] != 0);
        Lg[((stile * B_DIM + b) * 16 + s) * T_DIM + t2] = ok ? sum : NEG_INF;
    }
}

// ---------------------------------------------------------------------------
// Kernel 2b: per (s-tile, b): stage logits to LDS (async-to-LDS when the
// toolchain exposes it), softmax over t, then context = P @ V_b via WMMA.
// ---------------------------------------------------------------------------
__global__ __launch_bounds__(256)
void softctx_kernel(const float* __restrict__ Lg,
                    const __bf16* __restrict__ VT,
                    float* __restrict__ out) {
    __shared__ __align__(16) float  s_logit[16 * T_DIM];   // [s][t]
    __shared__ __align__(16) __bf16 s_p[16 * T_DIM];       // [s][t] bf16 probs

    const int tid   = threadIdx.x;
    const int lane  = tid & 31;
    const int wave  = tid >> 5;
    const int stile = blockIdx.x & 15;
    const int b     = blockIdx.x >> 4;
    const int sbase = stile * 16;

    // ---- stage 16x256 f32 logit tile; contiguous b128 async copies ----
    const float* __restrict__ Lrow = Lg + ((stile * B_DIM + b) * 16) * T_DIM;
    {
        const int base = tid * 16;                 // 4096 floats / 256 threads
        #pragma unroll
        for (int c = 0; c < 4; ++c)
            async_copy_b128(Lrow + base + c * 4, s_logit + base + c * 4);
        async_copy_wait();
    }
    __syncthreads();

    // ---- softmax over t; each wave owns 2 s-rows ----
    #pragma unroll
    for (int rr = 0; rr < 2; ++rr) {
        const int s = wave * 2 + rr;
        const float* row = s_logit + s * T_DIM;
        float mx = -3.4e38f;
        #pragma unroll
        for (int j = 0; j < 8; ++j) mx = fmaxf(mx, row[lane * 8 + j]);
        for (int off = 16; off > 0; off >>= 1)
            mx = fmaxf(mx, __shfl_xor(mx, off, 32));
        float sum = 0.0f;
        #pragma unroll
        for (int j = 0; j < 8; ++j) sum += fast_exp(row[lane * 8 + j] - mx);
        for (int off = 16; off > 0; off >>= 1)
            sum += __shfl_xor(sum, off, 32);
        const float rs = 1.0f / sum;
        #pragma unroll
        for (int j = 0; j < 8; ++j)
            s_p[s * T_DIM + lane * 8 + j] =
                (__bf16)(fast_exp(row[lane * 8 + j] - mx) * rs);
    }
    __syncthreads();

    // ---- context = P @ V_b, WMMA bf16; each wave: 2 N-tiles of 16 ----
    const __bf16* __restrict__ Vb = VT + b * D_DIM * T_DIM;
    const int arow   = lane & 15;
    const int abase0 = (lane < 16) ? 0 : 8;
    const int bbase  = (lane < 16) ? 0 : 8;
    #pragma unroll
    for (int half = 0; half < 2; ++half) {
        const int nt   = wave + half * 8;
        const int dcol = nt * 16 + (lane & 15);
        f32x8 accv = {};
        #pragma unroll
        for (int k0 = 0; k0 < T_DIM; k0 += 32) {
            ABv av, bv;
            av.q[0] = *(const uint4*)(s_p + arow * T_DIM + k0 + abase0);
            av.q[1] = *(const uint4*)(s_p + arow * T_DIM + k0 + abase0 + 16);
            bv.q[0] = *(const uint4*)(Vb + dcol * T_DIM + k0 + bbase);
            bv.q[1] = *(const uint4*)(Vb + dcol * T_DIM + k0 + bbase + 16);
            accv = __builtin_amdgcn_wmma_f32_16x16x32_bf16(
                false, av.v, false, bv.v, (short)0, accv, false, false);
        }
        const int mrow0 = (lane < 16) ? 0 : 8;
        #pragma unroll
        for (int r = 0; r < 8; ++r) {
            const int sg = sbase + mrow0 + r;
            out[(sg * B_DIM + b) * (2 * D_DIM) + dcol] = accv[r];
        }
    }
}

// ---------------------------------------------------------------------------
extern "C" void kernel_launch(void* const* d_in, const int* in_sizes, int n_in,
                              void* d_out, int out_size, void* d_ws, size_t ws_size,
                              hipStream_t stream) {
    const float* input  = (const float*)d_in[0];
    const float* memory = (const float*)d_in[1];
    const int*   mmask  = (const int*)d_in[2];
    const float* Wi     = (const float*)d_in[3];
    const float* Wm     = (const float*)d_in[4];
    const float* v      = (const float*)d_in[5];
    float*       out    = (float*)d_out;

    char* ws = (char*)d_ws;
    float*  ipf = (float*)(ws + 0);                          // 1 MB
    float*  mpf = (float*)(ws + (1u << 20));                 // 1 MB
    __bf16* Xbf = (__bf16*)(ws + (2u << 20));                // 1 MB
    __bf16* Mbf = (__bf16*)(ws + (3u << 20));                // 1 MB
    __bf16* WiT = (__bf16*)(ws + (4u << 20));                // 64 KB
    __bf16* WmT = (__bf16*)(ws + (4u << 20) + (64u << 10));  // 64 KB
    __bf16* VT  = (__bf16*)(ws + (4u << 20) + (128u << 10)); // 1 MB
    float*  Lg  = (float*)(ws + (6u << 20));                 // 2 MB logits

    prep_kernel<<<512, 256, 0, stream>>>(input, memory, Wi, Wm, out,
                                         Xbf, Mbf, WiT, WmT, VT);
    proj_kernel<<<512, 128, 0, stream>>>(Xbf, Mbf, WiT, WmT, ipf, mpf);
    logits_kernel<<<512, 256, 0, stream>>>(ipf, mpf, mmask, v, Lg);
    softctx_kernel<<<(T_DIM / 16) * B_DIM, 256, 0, stream>>>(Lg, VT, out);
}
